// LightClassifier_54202487275569
// MI455X (gfx1250) — matrix operand order repaired and checked by reading, not compiled, not run on hardware
//
#include <hip/hip_runtime.h>
#include <hip/hip_bf16.h>
#include <cstddef>

#define HD   128        // hidden width (Fout of every layer)
#define NEG  0.01f      // leaky-relu slope

typedef __attribute__((ext_vector_type(16))) _Float16 v16h;
typedef __attribute__((ext_vector_type(4)))  _Float16 v4h;
typedef __attribute__((ext_vector_type(8)))  float    v8f;

// ---------------------------------------------------------------------------
// Degree / normalization kernels
// ---------------------------------------------------------------------------
__global__ void deg_init_kernel(float* deg1, float* deg2, int N) {
    int i = blockIdx.x * blockDim.x + threadIdx.x;
    if (i < N) { deg1[i] = 1.0f; deg2[i] = 1.0f; }   // self-loop weight = 1
}

__global__ void deg_accum_kernel(const int* __restrict__ dst,
                                 const float* __restrict__ ew,
                                 float* deg1, float* deg2, int E) {
    int e = blockIdx.x * blockDim.x + threadIdx.x;
    if (e >= E) return;
    int d = dst[e];
    atomicAdd(&deg1[d], ew[e]);   // weighted in-degree (layer 1)
    atomicAdd(&deg2[d], 1.0f);    // unweighted in-degree (layer 2)
}

__global__ void deg_rsqrt_kernel(float* deg, int N) {
    int i = blockIdx.x * blockDim.x + threadIdx.x;
    if (i >= N) return;
    float d = deg[i];
    deg[i] = (d > 0.0f) ? rsqrtf(d) : 0.0f;
}

// ---------------------------------------------------------------------------
// WMMA GEMM:  out[N, 128] = A[N, K] @ W[128, K]^T   (+bias, leaky-relu optional)
// W is converted fp32->f16 ONCE per workgroup into LDS (row-major [128][K]);
// all 8 waves then feed v_wmma_f32_16x16x32_f16 from ds_load_b128, sharing W
// and removing per-iteration cvt work.  A is streamed fp32 from HBM and
// converted in registers (read exactly once).
// Fragment layouts used by this kernel (wave32):
//   A 16x32 f16 : lane L holds row L%16, K = (L/16)*16 + {0..15}
//   B 32x16 f16 : lane L holds col L%16, K = (L/16)*16 + {0..15}
//   C 16x16 f32 : lane L holds col L%16, rows (L/16)*8 + {0..7}
// ---------------------------------------------------------------------------
template<int K, bool BIAS_LRELU>
__global__ void gemm_wmma_kernel(const float* __restrict__ A,
                                 const float* __restrict__ W,
                                 const float* __restrict__ bias,
                                 float* __restrict__ out, int Nrows) {
    __shared__ __align__(32) _Float16 Wh[HD * K];

    // ---- cooperative W fp32 -> f16 staging (64-bit LDS stores) ----
    {
        const float4* W4  = (const float4*)W;
        v4h*          Wh4 = (v4h*)Wh;
        for (int p = threadIdx.x; p < HD * K / 4; p += blockDim.x) {
            float4 w = W4[p];
            v4h h4;
            h4[0] = (_Float16)w.x; h4[1] = (_Float16)w.y;
            h4[2] = (_Float16)w.z; h4[3] = (_Float16)w.w;
            Wh4[p] = h4;
        }
    }
    __syncthreads();

    const int lane = threadIdx.x & 31;
    const int wave = threadIdx.x >> 5;
    const int row0 = (blockIdx.x * (int)(blockDim.x >> 5) + wave) * 16;
    if (row0 >= Nrows) return;                 // wave-uniform: EXEC stays all-1s

    v8f acc[8];
    #pragma unroll
    for (int t = 0; t < 8; ++t)
        #pragma unroll
        for (int r = 0; r < 8; ++r) acc[t][r] = 0.0f;

    const int am = lane & 15;                  // A row inside tile
    const int ak = (lane >> 4) * 16;           // A K sub-offset (0 / 16)
    const int bn = lane & 15;                  // B column inside tile
    const int bk = (lane >> 4) * 16;           // B K sub-offset (0 / 16)

    for (int k0 = 0; k0 < K; k0 += 32) {
        // ---- A fragment (one 16-row strip, reused by all 8 col tiles) ----
        v16h a;
        const float* arow = A + (size_t)(row0 + am) * K + k0 + ak;
        #pragma unroll
        for (int i = 0; i < 8; ++i) a[i]     = (_Float16)arow[i];
        #pragma unroll
        for (int i = 0; i < 8; ++i) a[8 + i] = (_Float16)arow[8 + i];

        // ---- 8 column tiles of W from LDS (pre-converted f16) ----
        #pragma unroll
        for (int t = 0; t < 8; ++t) {
            v16h b = *(const v16h*)&Wh[(size_t)(t * 16 + bn) * K + k0 + bk];
            acc[t] = __builtin_amdgcn_wmma_f32_16x16x32_f16(
                         false, a, false, b, (short)0, acc[t], false, false);
        }
    }

    // ---- store C fragments ----
    const int n     = lane & 15;
    const int mbase = (lane >> 4) * 8;
    #pragma unroll
    for (int t = 0; t < 8; ++t) {
        const int col = t * 16 + n;
        float bv = BIAS_LRELU ? bias[col] : 0.0f;
        #pragma unroll
        for (int r = 0; r < 8; ++r) {
            float v = acc[t][r];
            if (BIAS_LRELU) { v += bv; v = (v > 0.0f) ? v : v * NEG; }
            out[(size_t)(row0 + mbase + r) * HD + col] = v;
        }
    }
}

// ---------------------------------------------------------------------------
// GCN aggregation:   out[i] = sum_j  norm(j->i) * h[j]   (incl. self loop)
// ---------------------------------------------------------------------------
__global__ void agg_self_kernel(const float* __restrict__ h,
                                const float* __restrict__ dinv,
                                float* __restrict__ out, long long total) {
    long long i = (long long)blockIdx.x * blockDim.x + threadIdx.x;
    if (i >= total) return;
    int node = (int)(i >> 7);                  // /128
    float d = dinv[node];
    out[i] = h[i] * d * d;                     // self-loop: dinv*1*dinv
}

__global__ void agg_edges_kernel(const float* __restrict__ h,
                                 const float* __restrict__ dinv,
                                 const int* __restrict__ src,
                                 const int* __restrict__ dst,
                                 const float* __restrict__ ew,   // null => 1.0
                                 float* __restrict__ out, int E) {
    long long t = (long long)blockIdx.x * blockDim.x + threadIdx.x;
    int e    = (int)(t >> 5);
    int lane = (int)(t & 31);
    if (e >= E) return;
    int s = src[e], d = dst[e];
    float w = ew ? ew[e] : 1.0f;
    float norm = dinv[s] * w * dinv[d];
    const float* hs = h + (size_t)s * HD;
    float*       od = out + (size_t)d * HD;
    #pragma unroll
    for (int j = 0; j < 4; ++j) {
        int f = lane + 32 * j;                 // coalesced 128B per quarter
        atomicAdd(&od[f], hs[f] * norm);       // non-returning global_atomic_add_f32
    }
}

__global__ void agg_finish_kernel(float* __restrict__ out,
                                  const float* __restrict__ bias,
                                  long long total) {
    long long i = (long long)blockIdx.x * blockDim.x + threadIdx.x;
    if (i >= total) return;
    int f = (int)(i & (HD - 1));
    float v = out[i] + bias[f];
    out[i] = (v > 0.0f) ? v : v * NEG;
}

// ---------------------------------------------------------------------------
// Classifier GEMV: out[n] = cls_w[0:128].x0[n] + cls_w[128:256].x2[n] + cls_b
// one wave per node, shuffle reduction (wave32)
// ---------------------------------------------------------------------------
__global__ void classifier_kernel(const float* __restrict__ x0,
                                  const float* __restrict__ x2,
                                  const float* __restrict__ cls_w,
                                  const float* __restrict__ cls_b,
                                  float* __restrict__ out, int N) {
    long long t = (long long)blockIdx.x * blockDim.x + threadIdx.x;
    int node = (int)(t >> 5);
    int lane = (int)(t & 31);
    if (node >= N) return;
    const float* r0 = x0 + (size_t)node * HD;
    const float* r2 = x2 + (size_t)node * HD;
    float s = 0.0f;
    #pragma unroll
    for (int j = 0; j < HD / 32; ++j) {
        int f = lane + 32 * j;
        s += r0[f] * cls_w[f] + r2[f] * cls_w[HD + f];
    }
    #pragma unroll
    for (int off = 16; off > 0; off >>= 1)
        s += __shfl_xor(s, off, 32);
    if (lane == 0) out[node] = s + cls_b[0];
}

// ---------------------------------------------------------------------------
// Launch
// ---------------------------------------------------------------------------
extern "C" void kernel_launch(void* const* d_in, const int* in_sizes, int n_in,
                              void* d_out, int out_size, void* d_ws, size_t ws_size,
                              hipStream_t stream) {
    const float* x      = (const float*)d_in[0];
    const int*   adj    = (const int*)  d_in[1];
    const float* weight = (const float*)d_in[2];
    const float* fc0_w  = (const float*)d_in[3];
    const float* fc0_b  = (const float*)d_in[4];
    const float* gnn1_w = (const float*)d_in[5];
    const float* gnn1_b = (const float*)d_in[6];
    const float* gnn2_w = (const float*)d_in[7];
    const float* gnn2_b = (const float*)d_in[8];
    const float* cls_w  = (const float*)d_in[9];
    const float* cls_b  = (const float*)d_in[10];

    const int Fin = in_sizes[3] / in_sizes[4];      // 32768 / 128 = 256
    const int N   = in_sizes[0] / Fin;              // 100000
    const int E   = in_sizes[2];                    // 1600000
    const int* src = adj;
    const int* dst = adj + E;

    // workspace layout (floats): [deg1/dinv1:N][deg2/dinv2:N][x0:N*H][h:N*H][agg:N*H]
    float* ws    = (float*)d_ws;
    float* dinv1 = ws;
    float* dinv2 = ws + (size_t)N;
    float* x0    = ws + (size_t)2 * N;
    float* hbuf  = x0   + (size_t)N * HD;
    float* abuf  = hbuf + (size_t)N * HD;

    const long long NH = (long long)N * HD;
    const int  TB      = 256;
    const int  gN      = (N + TB - 1) / TB;
    const int  gE      = (E + TB - 1) / TB;
    const int  gNH     = (int)((NH + TB - 1) / TB);
    const int  gE32    = (int)(((long long)E * 32 + TB - 1) / TB);
    const int  gN32    = (int)(((long long)N * 32 + TB - 1) / TB);
    const int  gGemm   = (N + 127) / 128;           // 8 waves x 16 rows / block

    // ---- degree normalizations (both layers) ----
    deg_init_kernel <<<gN, TB, 0, stream>>>(dinv1, dinv2, N);
    deg_accum_kernel<<<gE, TB, 0, stream>>>(dst, weight, dinv1, dinv2, E);
    deg_rsqrt_kernel<<<gN, TB, 0, stream>>>(dinv1, N);
    deg_rsqrt_kernel<<<gN, TB, 0, stream>>>(dinv2, N);

    // ---- fc0: x0 = lrelu(x @ fc0_w^T + fc0_b) ----
    gemm_wmma_kernel<256, true><<<gGemm, TB, 0, stream>>>(x, fc0_w, fc0_b, x0, N);

    // ---- GNN layer 1 (weighted edges) ----
    gemm_wmma_kernel<128, false><<<gGemm, TB, 0, stream>>>(x0, gnn1_w, nullptr, hbuf, N);
    agg_self_kernel  <<<gNH,  TB, 0, stream>>>(hbuf, dinv1, abuf, NH);
    agg_edges_kernel <<<gE32, TB, 0, stream>>>(hbuf, dinv1, src, dst, weight, abuf, E);
    agg_finish_kernel<<<gNH,  TB, 0, stream>>>(abuf, gnn1_b, NH);      // abuf = x1

    // ---- GNN layer 2 (unit edge weights) ----
    gemm_wmma_kernel<128, false><<<gGemm, TB, 0, stream>>>(abuf, gnn2_w, nullptr, hbuf, N);
    agg_self_kernel  <<<gNH,  TB, 0, stream>>>(hbuf, dinv2, abuf, NH); // overwrites x1 (dead)
    agg_edges_kernel <<<gE32, TB, 0, stream>>>(hbuf, dinv2, src, dst, nullptr, abuf, E);
    agg_finish_kernel<<<gNH,  TB, 0, stream>>>(abuf, gnn2_b, NH);      // abuf = x2

    // ---- classifier ----
    classifier_kernel<<<gN32, TB, 0, stream>>>(x0, abuf, cls_w, cls_b, (float*)d_out, N);
}
